// CascadePredictor_13675175870685
// MI455X (gfx1250) — matrix operand order repaired and checked by reading, not compiled, not run on hardware
//
#include <hip/hip_runtime.h>
#include <hip/hip_bf16.h>
#include <cstdint>

// ---------------------------------------------------------------------------
// CDNA5 / gfx1250 implementation of the GCN + 2-token-MHA link predictor.
//
// Math restructuring:
//   GCN layer:   out = dinv * ( scatter_add(g[src] -> dst) + g ),  g = (A@W+b)*dinv
//   MHA (seq=2, only token-0 output, summed over hidden):
//     per node:  Q = Z@Wq^T+bq, K = Z@Wk^T+bk   (WMMA GEMMs)
//                vd[n][h] = Z[n]·u_h + c_h      (u_h folds Wv and out_proj+sum)
//                qs[n][h] = scale * Q[n]_h · K[n]_h         (self score s00)
//     per edge:  s01_h = scale * Q[sp]_h · K[dp]_h
//                a1 = sigmoid(s01-s00);  o = vd0 + a1*(vd1-vd0)
//                out = sigmoid( sum_h o_h + bsum )
// ---------------------------------------------------------------------------

typedef __attribute__((ext_vector_type(16))) __bf16 v16bf;
typedef __attribute__((ext_vector_type(8)))  __bf16 v8bf;
typedef __attribute__((ext_vector_type(8)))  float  v8f;

// ---------------------------------- small utility kernels -------------------

__global__ void init_deg_kernel(float* deg, int n) {
  int i = blockIdx.x * blockDim.x + threadIdx.x;
  if (i < n) deg[i] = 1.0f;                       // self loop
}

__global__ void count_deg_kernel(const int* __restrict__ dst, float* __restrict__ deg, int E) {
  int i = blockIdx.x * blockDim.x + threadIdx.x;
  if (i < E) unsafeAtomicAdd(&deg[dst[i]], 1.0f);
}

__global__ void dinv_kernel(const float* __restrict__ deg, float* __restrict__ dinv, int n) {
  int i = blockIdx.x * blockDim.x + threadIdx.x;
  if (i < n) dinv[i] = rsqrtf(deg[i]);
}

__global__ void f32_to_bf16_kernel(const float* __restrict__ in, __bf16* __restrict__ out, int n) {
  int i = blockIdx.x * blockDim.x + threadIdx.x;
  if (i < n) out[i] = (__bf16)in[i];
}

// Pack row-major W[K,256] (or transposed W[256,K] when trans=1) into the WMMA
// B-operand lane layout: Bp[((kt*256 + n)*2 + sel)*16 + e]  holds  B[kt*32+sel*16+e][n]
// (lanes 0-15 of the wave read sel=0 -> K=kb..kb+15 ; lanes 16-31 read sel=1 -> K=kb+16..kb+31)
__global__ void pack_b_kernel(const float* __restrict__ W, __bf16* __restrict__ Bp,
                              int K, int trans) {
  int id = blockIdx.x * blockDim.x + threadIdx.x;
  if (id >= K * 256) return;
  int e   = id & 15;
  int sel = (id >> 4) & 1;
  int n   = (id >> 5) & 255;
  int kt  = id >> 13;
  int k = kt * 32 + sel * 16 + e;
  float v = trans ? W[(size_t)n * K + k] : W[(size_t)k * 256 + n];
  Bp[id] = (__bf16)v;
}

// ---------------------------------- WMMA GEMM -------------------------------
// C[M,256] = A[M,K](bf16,row-major) @ Bpacked + bias.
// One wave computes a 16x64 strip: 4 independent accumulators share one A tile
// per K-step (4 WMMAs / 10 b128 loads, no D->A/B hazards between them).
__global__ void gemm_bf16_wmma_kernel(const __bf16* __restrict__ A,
                                      const __bf16* __restrict__ Bp,
                                      const float* __restrict__ bias,
                                      float* __restrict__ C,
                                      int M, int K) {
  int gw   = (blockIdx.x * blockDim.x + threadIdx.x) >> 5;
  int lane = threadIdx.x & 31;
  int tiles = (M >> 4) << 2;                      // (M/16) * (256/64)
  if (gw >= tiles) return;
  int mt = gw >> 2;                               // 16-row tile
  int ng = gw & 3;                                // 64-col strip
  int half = lane >> 4;                           // 0: lanes 0-15, 1: lanes 16-31
  int lr   = lane & 15;

  const __bf16* ap = A + (size_t)(mt * 16 + lr) * K + half * 16;
  // B base for this strip: n = ng*64 + lr ; consecutive 16-col tiles are 512
  // elements apart, consecutive K-steps 8192 elements apart (imm offsets).
  const __bf16* bp = Bp + (((size_t)(ng * 64 + lr) * 2 + half) * 16);

  v8f acc0 = {}, acc1 = {}, acc2 = {}, acc3 = {};
  int ksteps = K >> 5;
  for (int kt = 0; kt < ksteps; ++kt) {
    // A lane layout used here (contiguous k, same as Bp): elems 0-15 = K=kb+16*half..kb+16*half+15
    v8bf a0 = *(const v8bf*)(ap);
    v8bf a1 = *(const v8bf*)(ap + 8);
    v16bf a = __builtin_shufflevector(a0, a1, 0,1,2,3,4,5,6,7,8,9,10,11,12,13,14,15);
    v16bf b0 = *(const v16bf*)(bp);
    v16bf b1 = *(const v16bf*)(bp + 512);
    v16bf b2 = *(const v16bf*)(bp + 1024);
    v16bf b3 = *(const v16bf*)(bp + 1536);
    acc0 = __builtin_amdgcn_wmma_f32_16x16x32_bf16(false, a, false, b0, (short)0, acc0, false, false);
    acc1 = __builtin_amdgcn_wmma_f32_16x16x32_bf16(false, a, false, b1, (short)0, acc1, false, false);
    acc2 = __builtin_amdgcn_wmma_f32_16x16x32_bf16(false, a, false, b2, (short)0, acc2, false, false);
    acc3 = __builtin_amdgcn_wmma_f32_16x16x32_bf16(false, a, false, b3, (short)0, acc3, false, false);
    ap += 32;
    bp += 8192;
  }
  // C/D layout: lane -> col = lr (within tile), row = half*8 + r
  float* crow = C + (size_t)(mt * 16 + half * 8) * 256 + ng * 64 + lr;
  float bi0 = bias[ng * 64 + lr];
  float bi1 = bias[ng * 64 + 16 + lr];
  float bi2 = bias[ng * 64 + 32 + lr];
  float bi3 = bias[ng * 64 + 48 + lr];
  #pragma unroll
  for (int r = 0; r < 8; ++r) {
    float* cr = crow + (size_t)r * 256;
    cr[0]  = acc0[r] + bi0;
    cr[16] = acc1[r] + bi1;
    cr[32] = acc2[r] + bi2;
    cr[48] = acc3[r] + bi3;
  }
}

// ---------------------------------- GCN aggregation -------------------------

// g = H * dinv[row] ; write g to G and also use it to initialize Acc (same buf as H).
__global__ void scale_init_kernel(float* __restrict__ HAcc, float* __restrict__ G,
                                  const float* __restrict__ dinv, int total4) {
  int i = blockIdx.x * blockDim.x + threadIdx.x;
  if (i >= total4) return;
  int n = i >> 6;                                 // 64 float4 per row of 256
  float dv = dinv[n];
  float4 h = ((const float4*)HAcc)[i];
  float4 g = make_float4(h.x * dv, h.y * dv, h.z * dv, h.w * dv);
  ((float4*)G)[i] = g;
  ((float4*)HAcc)[i] = g;
}

// 64 lanes per edge, 4 channels each: Acc[dst] += G[src]
__global__ void edge_agg_kernel(const int* __restrict__ src, const int* __restrict__ dst,
                                const float* __restrict__ G, float* __restrict__ Acc,
                                long long total) {
  long long gid = (long long)blockIdx.x * blockDim.x + threadIdx.x;
  if (gid >= total) return;
  int e  = (int)(gid >> 6);
  int c0 = ((int)gid & 63) << 2;
  int s = src[e], d = dst[e];
  float4 g = *(const float4*)(G + (size_t)s * 256 + c0);
  float* a = Acc + (size_t)d * 256 + c0;
  unsafeAtomicAdd(a + 0, g.x);
  unsafeAtomicAdd(a + 1, g.y);
  unsafeAtomicAdd(a + 2, g.z);
  unsafeAtomicAdd(a + 3, g.w);
}

// out = dinv * Acc (optionally relu), write bf16 copy (+ optional f32 copy)
__global__ void finalize_kernel(const float* __restrict__ Acc, const float* __restrict__ dinv,
                                __bf16* __restrict__ outBf, float* __restrict__ outF,
                                int relu, int total4) {
  int i = blockIdx.x * blockDim.x + threadIdx.x;
  if (i >= total4) return;
  int n = i >> 6;
  float dv = dinv[n];
  float4 a = ((const float4*)Acc)[i];
  float4 v = make_float4(a.x * dv, a.y * dv, a.z * dv, a.w * dv);
  if (relu) {
    v.x = fmaxf(v.x, 0.f); v.y = fmaxf(v.y, 0.f);
    v.z = fmaxf(v.z, 0.f); v.w = fmaxf(v.w, 0.f);
  }
  if (outF) ((float4*)outF)[i] = v;
  __bf16* ob = outBf + (size_t)i * 4;
  ob[0] = (__bf16)v.x; ob[1] = (__bf16)v.y; ob[2] = (__bf16)v.z; ob[3] = (__bf16)v.w;
}

// ---------------------------------- MHA folding -----------------------------

// wsum[i] = sum_j out_proj_w[j][i]
__global__ void wsum_kernel(const float* __restrict__ Wop, float* __restrict__ wsum) {
  int i = blockIdx.x * blockDim.x + threadIdx.x;
  if (i >= 256) return;
  float s = 0.f;
  for (int j = 0; j < 256; ++j) s += Wop[(size_t)j * 256 + i];
  wsum[i] = s;
}

// u[h*256+k] = sum_{t<64} wsum[h*64+t] * Wv[h*64+t][k], Wv = in_proj_w rows 512..767
__global__ void u_kernel(const float* __restrict__ Wip, const float* __restrict__ wsum,
                         float* __restrict__ u) {
  int i = blockIdx.x * blockDim.x + threadIdx.x;
  if (i >= 1024) return;
  int h = i >> 8, k = i & 255;
  float s = 0.f;
  for (int t = 0; t < 64; ++t)
    s += wsum[h * 64 + t] * Wip[(size_t)(512 + h * 64 + t) * 256 + k];
  u[i] = s;
}

// cvec[0..3] = c_h = sum wsum_h * bv_h ; cvec[4] = bsum = sum(out_proj_b)
__global__ void c_kernel(const float* __restrict__ bip, const float* __restrict__ bop,
                         const float* __restrict__ wsum, float* __restrict__ cvec) {
  int i = blockIdx.x * blockDim.x + threadIdx.x;
  if (i < 4) {
    float s = 0.f;
    for (int t = 0; t < 64; ++t) s += wsum[i * 64 + t] * bip[512 + i * 64 + t];
    cvec[i] = s;
  } else if (i == 4) {
    float s = 0.f;
    for (int j = 0; j < 256; ++j) s += bop[j];
    cvec[4] = s;
  }
}

// vd[n][h] = Z[n]·u_h + c_h
__global__ void vd_kernel(const float* __restrict__ Z, const float* __restrict__ u,
                          const float* __restrict__ cvec, float* __restrict__ vd, int NN) {
  int n = blockIdx.x * blockDim.x + threadIdx.x;
  if (n >= NN) return;
  const float* z = Z + (size_t)n * 256;
  float a0 = cvec[0], a1 = cvec[1], a2 = cvec[2], a3 = cvec[3];
  for (int k = 0; k < 256; ++k) {
    float zv = z[k];
    a0 += zv * u[k];
    a1 += zv * u[256 + k];
    a2 += zv * u[512 + k];
    a3 += zv * u[768 + k];
  }
  vd[(size_t)n * 4 + 0] = a0; vd[(size_t)n * 4 + 1] = a1;
  vd[(size_t)n * 4 + 2] = a2; vd[(size_t)n * 4 + 3] = a3;
}

// qs[n][h] = 0.125 * Q[n]_h · K[n]_h   (pre-scaled self attention score s00)
__global__ void qs_kernel(const float* __restrict__ Q, const float* __restrict__ Km,
                          float* __restrict__ qs, int NN) {
  int n = blockIdx.x * blockDim.x + threadIdx.x;
  if (n >= NN) return;
  const float* q = Q + (size_t)n * 256;
  const float* k = Km + (size_t)n * 256;
  #pragma unroll
  for (int h = 0; h < 4; ++h) {
    float s = 0.f;
    for (int t = 0; t < 64; ++t) s += q[h * 64 + t] * k[h * 64 + t];
    qs[(size_t)n * 4 + h] = 0.125f * s;
  }
}

// ---------------------------------- per-edge predictor ----------------------
// one wave per predicted edge
__global__ void edge_pred_kernel(const float* __restrict__ Q, const float* __restrict__ Km,
                                 const float* __restrict__ vd, const float* __restrict__ qs,
                                 const float* __restrict__ cvec,
                                 const int* __restrict__ sp, const int* __restrict__ dp,
                                 float* __restrict__ out, int E) {
  int wid  = (blockIdx.x * blockDim.x + threadIdx.x) >> 5;
  int lane = threadIdx.x & 31;
  if (wid >= E) return;
  int s = sp[wid], d = dp[wid];
  const float* q = Q  + (size_t)s * 256;
  const float* k = Km + (size_t)d * 256;
  float p[4] = {0.f, 0.f, 0.f, 0.f};
  #pragma unroll
  for (int j = 0; j < 8; ++j) {                   // k-index = 32*j + lane ; head = j/2
    int idx = j * 32 + lane;
    p[j >> 1] += q[idx] * k[idx];
  }
  #pragma unroll
  for (int off = 16; off > 0; off >>= 1) {
    p[0] += __shfl_xor(p[0], off);
    p[1] += __shfl_xor(p[1], off);
    p[2] += __shfl_xor(p[2], off);
    p[3] += __shfl_xor(p[3], off);
  }
  if (lane == 0) {
    float acc = cvec[4];                          // bsum
    #pragma unroll
    for (int h = 0; h < 4; ++h) {
      float s01 = 0.125f * p[h];
      float s00 = qs[(size_t)s * 4 + h];
      float a1 = 1.0f / (1.0f + __expf(s00 - s01));   // softmax over 2 keys
      float v0 = vd[(size_t)s * 4 + h];
      float v1 = vd[(size_t)d * 4 + h];
      acc += v0 + a1 * (v1 - v0);
    }
    out[wid] = 1.0f / (1.0f + __expf(-acc));
  }
}

// ---------------------------------- host launcher ---------------------------

static inline size_t alignup(size_t x) { return (x + 255) & ~(size_t)255; }

extern "C" void kernel_launch(void* const* d_in, const int* in_sizes, int n_in,
                              void* d_out, int out_size, void* d_ws, size_t ws_size,
                              hipStream_t stream) {
  const float* x   = (const float*)d_in[0];
  const int*   ei  = (const int*)d_in[1];
  const int*   eip = (const int*)d_in[2];
  const float* W1  = (const float*)d_in[3];
  const float* b1  = (const float*)d_in[4];
  const float* W2  = (const float*)d_in[5];
  const float* b2  = (const float*)d_in[6];
  const float* Wip = (const float*)d_in[7];
  const float* bip = (const float*)d_in[8];
  const float* Wop = (const float*)d_in[9];
  const float* bop = (const float*)d_in[10];
  float* out = (float*)d_out;

  const int NN = in_sizes[0] / 128;               // 50000
  const int E1 = in_sizes[1] / 2;                 // 1,600,000
  const int E2 = in_sizes[2] / 2;                 // 500,000
  const int* e_src = ei;           const int* e_dst = ei + E1;
  const int* p_src = eip;          const int* p_dst = eip + E2;

  // workspace layout
  char* ws = (char*)d_ws;
  size_t off = 0;
  auto take = [&](size_t bytes) -> void* { void* p = ws + off; off = alignup(off + bytes); return p; };
  float*  F1   = (float*) take((size_t)NN * 256 * 4);   // Hlin/Acc -> Zlin/Acc -> Q
  float*  F2   = (float*) take((size_t)NN * 256 * 4);   // G -> Z(f32) -> K
  __bf16* BFA  = (__bf16*)take((size_t)NN * 256 * 2);   // Xb -> Hb -> Zb
  float*  deg  = (float*) take((size_t)NN * 4);
  float*  dinv = (float*) take((size_t)NN * 4);
  __bf16* W1p  = (__bf16*)take((size_t)128 * 256 * 2);
  __bf16* W2p  = (__bf16*)take((size_t)256 * 256 * 2);
  __bf16* Wqp  = (__bf16*)take((size_t)256 * 256 * 2);
  __bf16* Wkp  = (__bf16*)take((size_t)256 * 256 * 2);
  float*  wsum = (float*) take(256 * 4);
  float*  uvec = (float*) take(1024 * 4);
  float*  cvec = (float*) take(8 * 4);
  float*  vd   = (float*) take((size_t)NN * 4 * 4);
  float*  qs   = (float*) take((size_t)NN * 4 * 4);
  (void)ws_size; (void)n_in; (void)out_size;

  const int B = 256;
  auto cdiv = [](long long a, long long b) { return (int)((a + b - 1) / b); };
  const int total4     = NN * 64;                 // float4 elements per [NN,256] buffer
  const int gemmBlocks = cdiv((long long)(NN / 16) * 4, 8); // 16x64 strips, 8 waves/block

  // ---- degrees / dinv -------------------------------------------------------
  init_deg_kernel<<<cdiv(NN, B), B, 0, stream>>>(deg, NN);
  count_deg_kernel<<<cdiv(E1, B), B, 0, stream>>>(e_dst, deg, E1);
  dinv_kernel<<<cdiv(NN, B), B, 0, stream>>>(deg, dinv, NN);

  // ---- weight packing (tiny, once per launch) -------------------------------
  pack_b_kernel<<<cdiv(128 * 256, B), B, 0, stream>>>(W1, W1p, 128, 0);
  pack_b_kernel<<<cdiv(256 * 256, B), B, 0, stream>>>(W2, W2p, 256, 0);
  pack_b_kernel<<<cdiv(256 * 256, B), B, 0, stream>>>(Wip,              Wqp, 256, 1); // q = z @ Wq^T
  pack_b_kernel<<<cdiv(256 * 256, B), B, 0, stream>>>(Wip + 256 * 256,  Wkp, 256, 1); // k = z @ Wk^T

  // ---- MHA folding constants ------------------------------------------------
  wsum_kernel<<<1, 256, 0, stream>>>(Wop, wsum);
  u_kernel<<<4, 256, 0, stream>>>(Wip, wsum, uvec);
  c_kernel<<<1, 32, 0, stream>>>(bip, bop, wsum, cvec);

  // ---- GCN layer 1 ----------------------------------------------------------
  f32_to_bf16_kernel<<<cdiv((long long)NN * 128, B), B, 0, stream>>>(x, BFA, NN * 128);
  gemm_bf16_wmma_kernel<<<gemmBlocks, B, 0, stream>>>(BFA, W1p, b1, F1, NN, 128);
  scale_init_kernel<<<cdiv(total4, B), B, 0, stream>>>(F1, F2, dinv, total4);
  edge_agg_kernel<<<cdiv((long long)E1 * 64, B), B, 0, stream>>>(e_src, e_dst, F2, F1, (long long)E1 * 64);
  finalize_kernel<<<cdiv(total4, B), B, 0, stream>>>(F1, dinv, BFA, nullptr, /*relu=*/1, total4);

  // ---- GCN layer 2 ----------------------------------------------------------
  gemm_bf16_wmma_kernel<<<gemmBlocks, B, 0, stream>>>(BFA, W2p, b2, F1, NN, 256);
  scale_init_kernel<<<cdiv(total4, B), B, 0, stream>>>(F1, F2, dinv, total4);
  edge_agg_kernel<<<cdiv((long long)E1 * 64, B), B, 0, stream>>>(e_src, e_dst, F2, F1, (long long)E1 * 64);
  finalize_kernel<<<cdiv(total4, B), B, 0, stream>>>(F1, dinv, BFA, F2, /*relu=*/0, total4); // F2 = Z(f32), BFA = Z(bf16)

  // ---- per-node attention precompute ---------------------------------------
  vd_kernel<<<cdiv(NN, B), B, 0, stream>>>(F2, uvec, cvec, vd, NN);     // uses Z before K overwrites F2
  gemm_bf16_wmma_kernel<<<gemmBlocks, B, 0, stream>>>(BFA, Wqp, bip,        F1, NN, 256); // Q
  gemm_bf16_wmma_kernel<<<gemmBlocks, B, 0, stream>>>(BFA, Wkp, bip + 256,  F2, NN, 256); // K
  qs_kernel<<<cdiv(NN, B), B, 0, stream>>>(F1, F2, qs, NN);

  // ---- per-edge prediction --------------------------------------------------
  edge_pred_kernel<<<cdiv((long long)E2 * 32, B), B, 0, stream>>>(F1, F2, vd, qs, cvec,
                                                                  p_src, p_dst, out, E2);
}
